// CrossAttentionBlock_16372415332889
// MI455X (gfx1250) — matrix operand order repaired and checked by reading, not compiled, not run on hardware
//
#include <hip/hip_runtime.h>

typedef __attribute__((ext_vector_type(16))) _Float16 v16h;
typedef __attribute__((ext_vector_type(8)))  _Float16 v8h;
typedef __attribute__((ext_vector_type(8)))  float    v8f;

// Problem dimensions
constexpr int Bn  = 32;
constexpr int Cn  = 768;
constexpr int Dn  = 512;
constexpr int NHn = 4;
constexpr int Hn  = 24;
constexpr int Wn  = 24;
constexpr int HDn = Cn / NHn;   // 192
constexpr int Fn  = Hn * Wn;    // 576
constexpr int C3  = 3 * Cn;     // 2304

enum { EPI_F16 = 0, EPI_F16_T_CB = 2, EPI_F32_RB_RES = 3, EPI_F32_RES = 4, EPI_F32 = 5 };

__device__ __forceinline__ v16h cat8(v8h lo, v8h hi) {
  return __builtin_shufflevector(lo, hi, 0,1,2,3,4,5,6,7,8,9,10,11,12,13,14,15);
}

__device__ __forceinline__ float blockReduceSum(float v, float* sb) {
  int t = threadIdx.x;
  sb[t] = v;
  __syncthreads();
  for (int s = 128; s > 0; s >>= 1) {
    if (t < s) sb[t] += sb[t + s];
    __syncthreads();
  }
  float r = sb[0];
  __syncthreads();
  return r;
}

// ---------------------------------------------------------------------------
// f32 -> f16 conversion
__global__ __launch_bounds__(256)
void cvt_kernel(const float* __restrict__ s, _Float16* __restrict__ d, int n) {
  int i = blockIdx.x * 256 + threadIdx.x;
  if (i < n) d[i] = (_Float16)s[i];
}

// ---------------------------------------------------------------------------
// Channel LayerNorm over C per (b,f) pixel; writes TRANSPOSED (B,F,C) f16.
__global__ __launch_bounds__(256)
void ln_ch_kernel(const float* __restrict__ x, const float* __restrict__ w,
                  const float* __restrict__ bb, _Float16* __restrict__ outT,
                  float eps) {
  __shared__ float sb[256];
  const int b = blockIdx.x / Fn;
  const int f = blockIdx.x % Fn;
  const float* xp = x + (long long)b * Cn * Fn + f;
  float s = 0.f, s2 = 0.f;
  for (int c = threadIdx.x; c < Cn; c += 256) {
    float v = xp[(long long)c * Fn];
    s += v; s2 += v * v;
  }
  float S  = blockReduceSum(s,  sb);
  float S2 = blockReduceSum(s2, sb);
  float mu   = S / Cn;
  float var  = S2 / Cn - mu * mu;
  float rstd = rsqrtf(var + eps);
  _Float16* op = outT + ((long long)b * Fn + f) * Cn;
  for (int c = threadIdx.x; c < Cn; c += 256) {
    float v = (xp[(long long)c * Fn] - mu) * rstd * w[c] + bb[c];
    op[c] = (_Float16)v;
  }
}

// ---------------------------------------------------------------------------
// Text LayerNorm (per batch row over D=512)
__global__ __launch_bounds__(256)
void txt_ln_kernel(const float* __restrict__ txt, const float* __restrict__ gw,
                   const float* __restrict__ gb, float* __restrict__ t) {
  __shared__ float sb[256];
  const int b = blockIdx.x;
  const float* xp = txt + (long long)b * Dn;
  float s = 0.f, s2 = 0.f;
  for (int d = threadIdx.x; d < Dn; d += 256) {
    float v = xp[d]; s += v; s2 += v * v;
  }
  float S  = blockReduceSum(s,  sb);
  float S2 = blockReduceSum(s2, sb);
  float mu   = S / Dn;
  float var  = S2 / Dn - mu * mu;
  float rstd = rsqrtf(var + 1e-5f);
  for (int d = threadIdx.x; d < Dn; d += 256)
    t[(long long)b * Dn + d] = (xp[d] - mu) * rstd * gw[d] + gb[d];
}

// ---------------------------------------------------------------------------
// sign(q_txt), sign(k_txt): tiny dot products against qkv_txt_w
__global__ __launch_bounds__(256)
void sign_kernel(const float* __restrict__ t, const float* __restrict__ wtxt,
                 float* __restrict__ s_q, float* __restrict__ s_k) {
  int tid = blockIdx.x * 256 + threadIdx.x;
  if (tid >= Bn * 2 * Cn) return;
  int b   = tid / (2 * Cn);
  int r   = tid % (2 * Cn);
  int sel = r / Cn;            // 0 = q, 1 = k
  int oi  = r % Cn;            // n*HD + h
  int n   = oi / HDn;
  int h   = oi % HDn;
  int row = n * 3 * HDn + sel * HDn + h;
  const float* tv = t + (long long)b * Dn;
  const float* wr = wtxt + (long long)row * Dn;
  float acc = 0.f;
  for (int d = 0; d < Dn; ++d) acc += tv[d] * wr[d];
  float s = acc / fmaxf(fabsf(acc), 1e-12f);
  (sel ? s_k : s_q)[(long long)b * Cn + oi] = s;
}

// ---------------------------------------------------------------------------
// attn rows: L2 norms over F, two softmaxes (args bounded in [-1,1]), summed.
__global__ __launch_bounds__(256)
void attn_kernel(const _Float16* __restrict__ qkv, const float* __restrict__ s_q,
                 const float* __restrict__ s_k, _Float16* __restrict__ asum) {
  __shared__ float sb[256];
  const int idx = blockIdx.x;                  // b*768 + (n*HD + h)
  const int b   = idx / (NHn * HDn);
  const int oi  = idx % (NHn * HDn);
  const _Float16* qrow = qkv + ((long long)b * C3 + oi) * Fn;
  const _Float16* krow = qrow + (long long)Cn * Fn;
  const float sq = s_q[(long long)b * Cn + oi];
  const float sk = s_k[(long long)b * Cn + oi];

  float aq = 0.f, ak = 0.f;
  for (int f = threadIdx.x; f < Fn; f += 256) {
    float qv = (float)qrow[f], kv = (float)krow[f];
    aq += qv * qv; ak += kv * kv;
  }
  float nq2 = blockReduceSum(aq, sb);
  float nk2 = blockReduceSum(ak, sb);
  float invq = 1.f / fmaxf(sqrtf(nq2), 1e-12f);
  float invk = 1.f / fmaxf(sqrtf(nk2), 1e-12f);

  float e1 = 0.f, e2 = 0.f;
  for (int f = threadIdx.x; f < Fn; f += 256) {
    e1 += __expf(sq * (float)krow[f] * invk);
    e2 += __expf(sk * (float)qrow[f] * invq);
  }
  float r1 = 1.f / blockReduceSum(e1, sb);
  float r2 = 1.f / blockReduceSum(e2, sb);

  _Float16* out = asum + (long long)idx * Fn;
  for (int f = threadIdx.x; f < Fn; f += 256) {
    float a1 = __expf(sq * (float)krow[f] * invk) * r1;
    float a2 = __expf(sk * (float)qrow[f] * invq) * r2;
    out[f] = (_Float16)(a1 + a2);
  }
}

// ---------------------------------------------------------------------------
// Depthwise 3x3 SAME conv + exact GELU; writes transposed (B,F,C) f16
__global__ __launch_bounds__(256)
void dwconv_gelu_kernel(const float* __restrict__ z1, const float* __restrict__ dw,
                        _Float16* __restrict__ gT) {
  long long tid = (long long)blockIdx.x * 256 + threadIdx.x;
  if (tid >= (long long)Bn * Cn * Fn) return;
  int f = (int)(tid % Fn);
  long long bc = tid / Fn;
  int c = (int)(bc % Cn);
  int b = (int)(bc / Cn);
  int hh = f / Wn, ww = f % Wn;
  const float* zp = z1 + ((long long)b * Cn + c) * Fn;
  const float* wp = dw + (long long)c * 9;
  float acc = 0.f;
#pragma unroll
  for (int dy = -1; dy <= 1; ++dy)
#pragma unroll
    for (int dx = -1; dx <= 1; ++dx) {
      int y = hh + dy, x = ww + dx;
      if (y >= 0 && y < Hn && x >= 0 && x < Wn)
        acc += wp[(dy + 1) * 3 + (dx + 1)] * zp[y * Wn + x];
    }
  float g = 0.5f * acc * (1.f + erff(acc * 0.70710678118f));
  gT[((long long)b * Fn + f) * Cn + c] = (_Float16)g;
}

// ---------------------------------------------------------------------------
// WMMA GEMM: D = A(M,K) * Bop(N,K)^T, f16 in / f32 accumulate.
// Each wave computes a 16x64 strip (4 accumulator tiles, A reused).
// Software-pipelined K loop: fragments for slice k+1 are loaded before the
// WMMAs of slice k, so loads overlap the matrix pipe instead of draining
// loadcnt to 0 in front of every v_wmma.
// A fragment per lane: M = lane&15, halves = K{kb+16a .. kb+16a+15} (a = lane>>4)
// B fragment per lane: N = lane&15, halves = K{kb+16a .. kb+16a+15}
// C/D: vgpr r, lane L -> row = r + 8*(L>=16), col = L&15
__global__ __launch_bounds__(256)
void wmma_gemm_kernel(const _Float16* __restrict__ A,
                      const _Float16* __restrict__ Bop,
                      void* __restrict__ Dst,
                      const float* __restrict__ bias,
                      const float* __restrict__ resid,
                      int M, int N, int K,
                      long long strideA,
                      long long strideB1, long long strideB2, int bdiv,
                      long long strideD, long long strideR,
                      int mode) {
  const int lane = threadIdx.x & 31;
  const int wave = threadIdx.x >> 5;
  const int tilesN4 = (N >> 4) >> 2;
  const int tilesM  = M >> 4;
  int wt = blockIdx.x * 8 + wave;
  if (wt >= tilesM * tilesN4) return;
  const int tm  = wt / tilesN4;
  const int tn0 = (wt % tilesN4) * 4;

  const long long zb = blockIdx.y;
  const _Float16* Ab = A + zb * strideA;
  const _Float16* Bb = Bop + (zb / bdiv) * strideB1 + (zb % bdiv) * strideB2;

  const int mrow  = lane & 15;
  const int koffA = (lane >> 4) << 4;   // 0 or 16
  const int ncol  = lane & 15;
  const int koffB = (lane >> 4) << 4;   // 0 or 16

  const _Float16* aptr = Ab + (long long)(tm * 16 + mrow) * K + koffA;
  const _Float16* bptr[4];
#pragma unroll
  for (int j = 0; j < 4; ++j)
    bptr[j] = Bb + (long long)((tn0 + j) * 16 + ncol) * K + koffB;

  // prologue: load K-slice 0
  v8h alo = *(const v8h*)(aptr);
  v8h ahi = *(const v8h*)(aptr + 8);
  v8h blo[4], bhi[4];
#pragma unroll
  for (int j = 0; j < 4; ++j) {
    blo[j] = *(const v8h*)(bptr[j]);
    bhi[j] = *(const v8h*)(bptr[j] + 8);
  }

  v8f acc[4] = {};

  for (int kb = 32; kb < K; kb += 32) {
    // issue loads for the NEXT K-slice first (overlaps with WMMAs below)
    v8h alo_n = *(const v8h*)(aptr + kb);
    v8h ahi_n = *(const v8h*)(aptr + kb + 8);
    v8h blo_n[4], bhi_n[4];
#pragma unroll
    for (int j = 0; j < 4; ++j) {
      blo_n[j] = *(const v8h*)(bptr[j] + kb);
      bhi_n[j] = *(const v8h*)(bptr[j] + kb + 8);
    }
    // consume CURRENT slice
    v16h a = cat8(alo, ahi);
#pragma unroll
    for (int j = 0; j < 4; ++j) {
      v16h bm = cat8(blo[j], bhi[j]);
      acc[j] = __builtin_amdgcn_wmma_f32_16x16x32_f16(
          false, a, false, bm, (short)0, acc[j], false, false);
    }
    // rotate buffers
    alo = alo_n; ahi = ahi_n;
#pragma unroll
    for (int j = 0; j < 4; ++j) { blo[j] = blo_n[j]; bhi[j] = bhi_n[j]; }
  }
  // epilogue: last K-slice
  {
    v16h a = cat8(alo, ahi);
#pragma unroll
    for (int j = 0; j < 4; ++j) {
      v16h bm = cat8(blo[j], bhi[j]);
      acc[j] = __builtin_amdgcn_wmma_f32_16x16x32_f16(
          false, a, false, bm, (short)0, acc[j], false, false);
    }
  }

  const int rbase = (lane >> 4) << 3;  // 0 or 8
  if (mode == EPI_F16) {
    _Float16* D = (_Float16*)Dst + zb * strideD;
#pragma unroll
    for (int j = 0; j < 4; ++j) {
      int col = (tn0 + j) * 16 + ncol;
#pragma unroll
      for (int r = 0; r < 8; ++r) {
        int row = tm * 16 + rbase + r;
        D[(long long)row * N + col] = (_Float16)acc[j][r];
      }
    }
  } else if (mode == EPI_F16_T_CB) {
    _Float16* D = (_Float16*)Dst + zb * strideD;
#pragma unroll
    for (int j = 0; j < 4; ++j) {
      int col = (tn0 + j) * 16 + ncol;
      float cb = bias[col];
#pragma unroll
      for (int r = 0; r < 8; ++r) {
        int row = tm * 16 + rbase + r;
        D[(long long)col * M + row] = (_Float16)(acc[j][r] + cb);
      }
    }
  } else {
    float* D = (float*)Dst + zb * strideD;
#pragma unroll
    for (int j = 0; j < 4; ++j) {
      int col = (tn0 + j) * 16 + ncol;
#pragma unroll
      for (int r = 0; r < 8; ++r) {
        int row = tm * 16 + rbase + r;
        float v = acc[j][r];
        if (mode == EPI_F32_RB_RES) v += bias[row];
        if (mode == EPI_F32_RB_RES || mode == EPI_F32_RES)
          v += resid[zb * strideR + (long long)row * N + col];
        D[(long long)row * N + col] = v;
      }
    }
  }
}

// ---------------------------------------------------------------------------
extern "C" void kernel_launch(void* const* d_in, const int* in_sizes, int n_in,
                              void* d_out, int out_size, void* d_ws, size_t ws_size,
                              hipStream_t stream) {
  const float* img_feats  = (const float*)d_in[0];
  const float* txt_feats  = (const float*)d_in[1];
  const float* fn_w       = (const float*)d_in[2];
  const float* fn_b       = (const float*)d_in[3];
  const float* gn_w       = (const float*)d_in[4];
  const float* gn_b       = (const float*)d_in[5];
  const float* qkv_img_w  = (const float*)d_in[6];
  const float* qkv_txt_w  = (const float*)d_in[7];
  const float* o_img_w    = (const float*)d_in[8];
  const float* o_img_b    = (const float*)d_in[9];
  const float* o_img2_w   = (const float*)d_in[10];
  const float* o_img2_b   = (const float*)d_in[11];
  const float* ffn_norm_w = (const float*)d_in[12];
  const float* ffn_norm_b = (const float*)d_in[13];
  const float* fc1_w      = (const float*)d_in[14];
  const float* dw_w       = (const float*)d_in[15];
  const float* fc2_w      = (const float*)d_in[16];

  char* ws = (char*)d_ws;
  size_t off = 0;
  auto alloc = [&](size_t bytes) -> void* {
    void* p = ws + off;
    off = (off + bytes + 255) & ~(size_t)255;
    return p;
  };

  _Float16* xT     = (_Float16*)alloc((size_t)Bn * Fn * Cn * 2);   // LN(img) transposed (B,F,C)
  _Float16* qkv    = (_Float16*)alloc((size_t)Bn * C3 * Fn * 2);   // (B,2304,576)
  float*    t_ln   = (float*)   alloc((size_t)Bn * Dn * 4);
  float*    s_q    = (float*)   alloc((size_t)Bn * Cn * 4);
  float*    s_k    = (float*)   alloc((size_t)Bn * Cn * 4);
  _Float16* asum   = (_Float16*)alloc((size_t)Bn * NHn * HDn * Fn * 2);
  _Float16* img_h  = (_Float16*)alloc((size_t)Bn * Cn * HDn * 2);  // (B,768,192)
  _Float16* yT     = (_Float16*)alloc((size_t)Bn * Fn * Cn * 2);   // (B,F,C)
  float*    outb   = (float*)   alloc((size_t)Bn * Cn * Fn * 4);   // residual "out"
  _Float16* zT     = (_Float16*)alloc((size_t)Bn * Fn * Cn * 2);
  float*    z1     = (float*)   alloc((size_t)Bn * Cn * Fn * 4);
  _Float16* gT     = (_Float16*)alloc((size_t)Bn * Fn * Cn * 2);
  _Float16* wqkv_h = (_Float16*)alloc((size_t)C3 * Cn * 2);
  _Float16* wo2_h  = (_Float16*)alloc((size_t)Fn * HDn * 2);
  _Float16* wo_h   = (_Float16*)alloc((size_t)Cn * Cn * 2);
  _Float16* wfc1_h = (_Float16*)alloc((size_t)Cn * Cn * 2);
  _Float16* wfc2_h = (_Float16*)alloc((size_t)Cn * Cn * 2);

  auto cvt = [&](const float* s, _Float16* d, int n) {
    cvt_kernel<<<(n + 255) / 256, 256, 0, stream>>>(s, d, n);
  };
  auto gemm = [&](const _Float16* A, const _Float16* Bp, void* D,
                  const float* bias, const float* resid,
                  int M, int N, int K,
                  long long sA, long long sB1, long long sB2, int bdiv,
                  long long sD, long long sR, int mode, int nz) {
    int waveTiles = (M / 16) * (N / 64);
    dim3 grid((waveTiles + 7) / 8, nz);
    wmma_gemm_kernel<<<grid, 256, 0, stream>>>(A, Bp, D, bias, resid, M, N, K,
                                               sA, sB1, sB2, bdiv, sD, sR, mode);
  };

  // Weight conversion to f16
  cvt(qkv_img_w, wqkv_h, C3 * Cn);
  cvt(o_img2_w,  wo2_h,  Fn * HDn);
  cvt(o_img_w,   wo_h,   Cn * Cn);
  cvt(fc1_w,     wfc1_h, Cn * Cn);
  cvt(fc2_w,     wfc2_h, Cn * Cn);

  // x = LayerNorm_C(img_feats) -> xT (B,F,C)
  ln_ch_kernel<<<Bn * Fn, 256, 0, stream>>>(img_feats, fn_w, fn_b, xT, 1e-6f);
  // txt LayerNorm + signs
  txt_ln_kernel<<<Bn, 256, 0, stream>>>(txt_feats, gn_w, gn_b, t_ln);
  sign_kernel<<<(Bn * 2 * Cn + 255) / 256, 256, 0, stream>>>(t_ln, qkv_txt_w, s_q, s_k);

  // G1: qkv = Wqkv(2304x768) @ x(768x576)  -> (B,2304,576) f16
  gemm(wqkv_h, xT, qkv, nullptr, nullptr, C3, Fn, Cn,
       0, (long long)Fn * Cn, 0, 1, (long long)C3 * Fn, 0, EPI_F16, Bn);

  // attention rows -> asum (B*NH, HD, F) f16
  attn_kernel<<<Bn * NHn * HDn, 256, 0, stream>>>(qkv, s_q, s_k, asum);

  // G2: img = asum(192x576) @ v^T(576x192) per (b,n); v rows live at qkv[b, 1536+n*HD+g, :]
  gemm(asum, qkv + (size_t)2 * Cn * Fn, img_h, nullptr, nullptr,
       HDn, HDn, Fn,
       (long long)HDn * Fn,
       (long long)C3 * Fn, (long long)HDn * Fn, NHn,
       (long long)HDn * HDn, 0, EPI_F16, Bn * NHn);

  // G3: y = img(768x192) @ o_img2_w^T(192x576) + b  -> yT (B,F,C) f16 (transposed store)
  gemm(img_h, wo2_h, yT, o_img2_b, nullptr,
       Cn, Fn, HDn,
       (long long)Cn * HDn, 0, 0, 1,
       (long long)Fn * Cn, 0, EPI_F16_T_CB, Bn);

  // G4: out = o_img_w(768x768) @ y + o_img_b + img_feats  -> outb f32
  gemm(wo_h, yT, outb, o_img_b, img_feats,
       Cn, Fn, Cn,
       0, (long long)Fn * Cn, 0, 1,
       (long long)Cn * Fn, (long long)Cn * Fn, EPI_F32_RB_RES, Bn);

  // z = LayerNorm_C(out) -> zT (B,F,C)
  ln_ch_kernel<<<Bn * Fn, 256, 0, stream>>>(outb, ffn_norm_w, ffn_norm_b, zT, 1e-6f);

  // G5: z1 = fc1_w @ z  -> f32 (B,C,F)
  gemm(wfc1_h, zT, z1, nullptr, nullptr,
       Cn, Fn, Cn,
       0, (long long)Fn * Cn, 0, 1,
       (long long)Cn * Fn, 0, EPI_F32, Bn);

  // depthwise conv 3x3 + exact GELU -> gT (B,F,C) f16
  dwconv_gelu_kernel<<<(int)(((long long)Bn * Cn * Fn + 255) / 256), 256, 0, stream>>>(
      z1, dw_w, gT);

  // G6: final = out + fc2_w @ g  -> d_out f32 (B,C,H,W)
  gemm(wfc2_h, gT, d_out, nullptr, outb,
       Cn, Fn, Cn,
       0, (long long)Fn * Cn, 0, 1,
       (long long)Cn * Fn, (long long)Cn * Fn, EPI_F32_RES, Bn);
}
